// ObsEncoder_60189671686452
// MI455X (gfx1250) — compile-verified, matrix operand order checked by reading, not run on hardware
//
#include <hip/hip_runtime.h>

#define HID      256
#define OBS_DIM  221
#define BM       32      // batch rows per workgroup
#define NTHREADS 256     // 8 wave32s

typedef __attribute__((ext_vector_type(16))) __bf16 v16bf;
typedef __attribute__((ext_vector_type(8)))  float  v8f;

union FragU {
  uint4 q[2];
  v16bf v;
};

// ---- workspace layout (bf16 elements) ----
enum : int {
  OFF_WT_SELF = 0,
  OFF_WT_SEEK = OFF_WT_SELF + 256 * 32,
  OFF_WT_HIDE = OFF_WT_SEEK + 256 * 32,
  OFF_WT_BOX  = OFF_WT_HIDE + 256 * 32,
  OFF_WT_RAMP = OFF_WT_BOX  + 256 * 32,
  OFF_CT_ADV  = OFF_WT_RAMP + 256 * 32,
  OFF_CT_GOOD = OFF_CT_ADV  + 256 * 256,
  OFF_CT_BOX  = OFF_CT_GOOD + 256 * 256,
  OFF_CT_RAMP = OFF_CT_BOX  + 256 * 256,
  OFF_WT_FC   = OFF_CT_RAMP + 256 * 256,
  OFF_WT_E1   = OFF_WT_FC   + 256 * 256,
  OFF_WT_E2   = OFF_WT_E1   + 256 * 1280,
  WS_ELEMS    = OFF_WT_E2   + 256 * 256
};

#define ACT_NONE     0
#define ACT_TANH     1
#define OUT_BF16_LDS 0
#define OUT_F32_LDS  1
#define OUT_F32_GLOB 2

__device__ __forceinline__ unsigned short f32_to_bf16(float f) {
  unsigned u = __float_as_uint(f);
  u += 0x7fffu + ((u >> 16) & 1u);   // round-to-nearest-even
  return (unsigned short)(u >> 16);
}
__device__ __forceinline__ float bf16_to_f32(unsigned short h) {
  return __uint_as_float(((unsigned)h) << 16);
}
// branchless tanh: tanh(x) = 1 - 2*rcp(e^{2x}+1)
// one v_exp_f32 + v_rcp_f32 + 2 VALU; no IEEE div expansion, no EXEC divergence
__device__ __forceinline__ float fast_tanh(float x) {
  float xc = fminf(fmaxf(x, -12.0f), 12.0f);
  float e  = __expf(2.0f * xc);
  return 1.0f - 2.0f * __builtin_amdgcn_rcpf(e + 1.0f);
}

// Fused-tile LDS working set (~315 KB, fits 320 KB/WGP; 1 WG = 8 waves per WGP)
struct SMem {
  unsigned short x[288 * 32];        //  18432 B : padded entity inputs (WMMA A)
  unsigned short embg[288 * 256];    // 147456 B : current group embeddings
  unsigned short selfemb[BM * 256];  //  16384 B : emb_self
  unsigned short cat[BM * 1280];     //  81920 B : [gi | vi_seek | vi_hide | vi_box | vi_ramp]
  unsigned short h[BM * 256];        //  16384 B : hidden
  float          q[BM * 256];        //  32768 B : attention query (f32)
  float          beta[BM * 12];      //   1536 B : scores -> alphas
};

// ---- prep: fp32 W[K][N] -> bf16 Wt[n*Kpad + k] (transposed, K zero-padded) ----
__global__ void prep_wt(const float* __restrict__ W, unsigned short* __restrict__ Wt,
                        int K, int N, int Kpad) {
  int idx = blockIdx.x * blockDim.x + threadIdx.x;
  if (idx >= N * Kpad) return;
  int k = idx % Kpad;
  int n = idx / Kpad;
  float v = (k < K) ? W[(size_t)k * N + n] : 0.0f;
  Wt[idx] = f32_to_bf16(v);
}

// One wave computes a 32-wide N slice of D(Mrows x 256) = A(Mrows x K) @ B(K x 256).
// Per k-step: 2 A frags (LDS) x 2 B frags (global, L2-resident) -> 4 WMMAs,
// so each fragment is reused twice (32 FLOP per L2 byte).
// A: LDS bf16 row-major (stride Astride). Bt: global bf16, pre-transposed [256][K].
// Mrows must be a multiple of 32.
template <int K, int ACT, int OUTM, bool HASB>
__device__ __forceinline__ void wave_matmul(
    const unsigned short* __restrict__ Bt,
    const unsigned short* As, int Astride, int Mrows,
    const float* __restrict__ bias,
    unsigned short* Dbf, float* Df, float* __restrict__ Dg,
    int Dstride, int dcol0, int lane, int wave)
{
  const int ncolbase = wave * 32;
  const int r  = lane & 15;           // M row (A) / N col (B) within tile
  const int c0 = (lane >> 4) << 3;    // per-lane K-chunk base: 0 or 8 (+16 for 2nd chunk)

  // bias per N column handled by this lane (same for both C/D row halves)
  float bv[2] = {0.0f, 0.0f};
  if constexpr (HASB) {
#pragma unroll
    for (int nt = 0; nt < 2; ++nt) bv[nt] = bias[ncolbase + nt * 16 + r];
  }

  FragU bK[2];                        // persistent B when K==32 (embeddings)
  if constexpr (K == 32) {
#pragma unroll
    for (int nt = 0; nt < 2; ++nt) {
      const unsigned short* bp = Bt + (size_t)(ncolbase + nt * 16 + r) * K + c0;
      bK[nt].q[0] = *(const uint4*)(bp);
      bK[nt].q[1] = *(const uint4*)(bp + 16);
    }
  }

  for (int mt0 = 0; mt0 < Mrows; mt0 += 32) {
    v8f z = {0.f, 0.f, 0.f, 0.f, 0.f, 0.f, 0.f, 0.f};
    v8f acc[2][2];
#pragma unroll
    for (int mi = 0; mi < 2; ++mi)
#pragma unroll
      for (int nt = 0; nt < 2; ++nt) acc[mi][nt] = z;

    for (int kb = 0; kb < K; kb += 32) {
      FragU a[2];
#pragma unroll
      for (int mi = 0; mi < 2; ++mi) {
        const unsigned short* ap =
            As + (size_t)(mt0 + mi * 16 + r) * Astride + kb + c0;
        a[mi].q[0] = *(const uint4*)(ap);
        a[mi].q[1] = *(const uint4*)(ap + 16);
      }
      FragU b[2];
      if constexpr (K == 32) {
        b[0] = bK[0]; b[1] = bK[1];
      } else {
#pragma unroll
        for (int nt = 0; nt < 2; ++nt) {
          const unsigned short* bp =
              Bt + (size_t)(ncolbase + nt * 16 + r) * K + kb + c0;
          b[nt].q[0] = *(const uint4*)(bp);
          b[nt].q[1] = *(const uint4*)(bp + 16);
        }
      }
#pragma unroll
      for (int mi = 0; mi < 2; ++mi)
#pragma unroll
        for (int nt = 0; nt < 2; ++nt)
          acc[mi][nt] = __builtin_amdgcn_wmma_f32_16x16x32_bf16(
              false, a[mi].v, false, b[nt].v, (short)0, acc[mi][nt], false, false);
    }
    // epilogue: bias + act, scatter per C/D lane layout (lane<16: m=i, else m=8+i)
#pragma unroll
    for (int mi = 0; mi < 2; ++mi) {
      const int mrow0 = mt0 + mi * 16 + ((lane >> 4) << 3);
#pragma unroll
      for (int nt = 0; nt < 2; ++nt) {
        const int nglob = ncolbase + nt * 16 + r;
#pragma unroll
        for (int i = 0; i < 8; ++i) {
          float v = acc[mi][nt][i] + bv[nt];
          if constexpr (ACT == ACT_TANH) v = fast_tanh(v);
          const int rr = mrow0 + i;
          if constexpr (OUTM == OUT_BF16_LDS)
            Dbf[rr * Dstride + dcol0 + nglob] = f32_to_bf16(v);
          else if constexpr (OUTM == OUT_F32_LDS)
            Df[rr * Dstride + dcol0 + nglob] = v;
          else
            Dg[(size_t)rr * HID + nglob] = v;
        }
      }
    }
  }
}

// Stage BM batch rows of one entity group into a zero-padded (BM*nent) x 32 bf16 A matrix.
__device__ __forceinline__ void stage_group(const float* __restrict__ inp, int batch0,
                                            int nent, int incols, int srcoff,
                                            int entstride, unsigned short* dst) {
  const int total = BM * nent * 32;
  for (int idx = threadIdx.x; idx < total; idx += NTHREADS) {
    int col = idx & 31;
    int row = idx >> 5;
    int b = row / nent;
    int j = row - b * nent;
    float v = 0.0f;
    if (col < incols)
      v = inp[(size_t)(batch0 + b) * OBS_DIM + srcoff + j * entstride + col];
    dst[idx] = f32_to_bf16(v);
  }
}

__global__ void __launch_bounds__(NTHREADS) obs_encoder_wmma(
    const float* __restrict__ inp, const unsigned short* __restrict__ ws,
    const float* __restrict__ b_self_, const float* __restrict__ b_seek_,
    const float* __restrict__ b_hide_, const float* __restrict__ b_box_,
    const float* __restrict__ b_ramp_, const float* __restrict__ b_fc_,
    const float* __restrict__ b_e1_,  const float* __restrict__ b_e2_,
    float* __restrict__ out)
{
  __shared__ SMem sm;
  const int tid = threadIdx.x, lane = tid & 31, wave = tid >> 5;
  const int batch0 = blockIdx.x * BM;

  // emb_self = tanh(self_x @ W_self + b)  (K padded 10->32)
  stage_group(inp, batch0, 1, 10, 0, 10, sm.x);
  __syncthreads();
  wave_matmul<32, ACT_TANH, OUT_BF16_LDS, true>(
      ws + OFF_WT_SELF, sm.x, 32, BM, b_self_,
      sm.selfemb, nullptr, nullptr, 256, 0, lane, wave);
  __syncthreads();
  // gi = tanh(emb_self @ W_fc + b) -> cat[:, 0:256]
  wave_matmul<256, ACT_TANH, OUT_BF16_LDS, true>(
      ws + OFF_WT_FC, sm.selfemb, 256, BM, b_fc_,
      sm.cat, nullptr, nullptr, 1280, 0, lane, wave);

  const int    nents[4]   = {3, 4, 9, 2};
  const int    srcoffs[4] = {10, 40, 80, 197};
  const int    estr[4]    = {10, 10, 12, 12};
  const int    wtoff[4]   = {OFF_WT_SEEK, OFF_WT_HIDE, OFF_WT_BOX, OFF_WT_RAMP};
  const int    ctoff[4]   = {OFF_CT_ADV, OFF_CT_GOOD, OFF_CT_BOX, OFF_CT_RAMP};
  const float* gbias[4]   = {b_seek_, b_hide_, b_box_, b_ramp_};

  for (int g = 0; g < 4; ++g) {
    const int ne = nents[g];
    __syncthreads();
    stage_group(inp, batch0, ne, (g < 2) ? 10 : 12, srcoffs[g], estr[g], sm.x);
    __syncthreads();
    // group embeddings (BM*ne rows, multiple of 32) and attention query
    wave_matmul<32, ACT_TANH, OUT_BF16_LDS, true>(
        ws + wtoff[g], sm.x, 32, BM * ne, gbias[g],
        sm.embg, nullptr, nullptr, 256, 0, lane, wave);
    wave_matmul<256, ACT_NONE, OUT_F32_LDS, false>(
        ws + ctoff[g], sm.selfemb, 256, BM, nullptr,
        nullptr, sm.q, nullptr, 256, 0, lane, wave);
    __syncthreads();
    // beta[b,j] = q[b,:] . emb[b,j,:]  -- 8 lanes cooperate per pair
    {
      const int sub = tid & 7;
      for (int p = tid >> 3; p < BM * ne; p += NTHREADS / 8) {
        const int b = p / ne;
        const float* qr = sm.q + b * 256;
        const unsigned short* er = sm.embg + p * 256;
        float s = 0.0f;
        const int h0 = sub * 32;
        for (int hh = h0; hh < h0 + 32; ++hh) s += qr[hh] * bf16_to_f32(er[hh]);
        s += __shfl_xor(s, 1, 32);
        s += __shfl_xor(s, 2, 32);
        s += __shfl_xor(s, 4, 32);
        if (sub == 0) sm.beta[b * 12 + (p - b * ne)] = s;
      }
    }
    __syncthreads();
    // softmax over entities (one thread per batch row)
    if (tid < BM) {
      float mx = -1e30f;
      for (int j = 0; j < ne; ++j) mx = fmaxf(mx, sm.beta[tid * 12 + j]);
      float e[9], ssum = 0.0f;
      for (int j = 0; j < ne; ++j) { e[j] = __expf(sm.beta[tid * 12 + j] - mx); ssum += e[j]; }
      float inv = __builtin_amdgcn_rcpf(ssum);
      for (int j = 0; j < ne; ++j) sm.beta[tid * 12 + j] = e[j] * inv;
    }
    __syncthreads();
    // vi[b,:] = sum_j alpha[b,j] * emb[b,j,:] -> cat slice
    const int ccol = 256 * (1 + g);
    for (int o = tid; o < BM * 256; o += NTHREADS) {
      int b = o >> 8, hh = o & 255;
      float s = 0.0f;
      for (int j = 0; j < ne; ++j)
        s += sm.beta[b * 12 + j] * bf16_to_f32(sm.embg[(b * ne + j) * 256 + hh]);
      sm.cat[b * 1280 + ccol + hh] = f32_to_bf16(s);
    }
  }
  __syncthreads();
  // h = tanh(cat @ W_e1 + b_e1), K=1280
  wave_matmul<1280, ACT_TANH, OUT_BF16_LDS, true>(
      ws + OFF_WT_E1, sm.cat, 1280, BM, b_e1_,
      sm.h, nullptr, nullptr, 256, 0, lane, wave);
  __syncthreads();
  // out = tanh(h @ W_e2 + b_e2) -> global fp32
  wave_matmul<256, ACT_TANH, OUT_F32_GLOB, true>(
      ws + OFF_WT_E2, sm.h, 256, BM, b_e2_,
      nullptr, nullptr, out + (size_t)batch0 * HID, 256, 0, lane, wave);
}

extern "C" void kernel_launch(void* const* d_in, const int* in_sizes, int n_in,
                              void* d_out, int out_size, void* d_ws, size_t ws_size,
                              hipStream_t stream) {
  const float* inp    = (const float*)d_in[0];
  const float* W_self = (const float*)d_in[1];
  const float* b_self = (const float*)d_in[2];
  const float* W_seek = (const float*)d_in[3];
  const float* b_seek = (const float*)d_in[4];
  const float* W_hide = (const float*)d_in[5];
  const float* b_hide = (const float*)d_in[6];
  const float* W_box  = (const float*)d_in[7];
  const float* b_box  = (const float*)d_in[8];
  const float* W_ramp = (const float*)d_in[9];
  const float* b_ramp = (const float*)d_in[10];
  const float* C_adv  = (const float*)d_in[11];
  const float* C_good = (const float*)d_in[12];
  const float* C_box  = (const float*)d_in[13];
  const float* C_ramp = (const float*)d_in[14];
  const float* W_fc   = (const float*)d_in[15];
  const float* b_fc   = (const float*)d_in[16];
  const float* W_e1   = (const float*)d_in[17];
  const float* b_e1   = (const float*)d_in[18];
  const float* W_e2   = (const float*)d_in[19];
  const float* b_e2   = (const float*)d_in[20];

  unsigned short* ws = (unsigned short*)d_ws;
  const int B = in_sizes[0] / OBS_DIM;

  auto prep = [&](const float* W, int off, int K, int N, int Kpad) {
    int total = N * Kpad;
    prep_wt<<<(total + 255) / 256, 256, 0, stream>>>(W, ws + off, K, N, Kpad);
  };
  prep(W_self, OFF_WT_SELF, 10, 256, 32);
  prep(W_seek, OFF_WT_SEEK, 10, 256, 32);
  prep(W_hide, OFF_WT_HIDE, 10, 256, 32);
  prep(W_box,  OFF_WT_BOX,  12, 256, 32);
  prep(W_ramp, OFF_WT_RAMP, 12, 256, 32);
  prep(C_adv,  OFF_CT_ADV,  256, 256, 256);
  prep(C_good, OFF_CT_GOOD, 256, 256, 256);
  prep(C_box,  OFF_CT_BOX,  256, 256, 256);
  prep(C_ramp, OFF_CT_RAMP, 256, 256, 256);
  prep(W_fc,   OFF_WT_FC,   256, 256, 256);
  prep(W_e1,   OFF_WT_E1,   1280, 256, 1280);
  prep(W_e2,   OFF_WT_E2,   256, 256, 256);

  obs_encoder_wmma<<<B / BM, NTHREADS, 0, stream>>>(
      inp, ws, b_self, b_seek, b_hide, b_box, b_ramp, b_fc, b_e1, b_e2,
      (float*)d_out);
}
